// NAFBlock_42640435314715
// MI455X (gfx1250) — matrix operand order repaired and checked by reading, not compiled, not run on hardware
//
#include <hip/hip_runtime.h>
#include <hip/hip_bf16.h>

typedef __attribute__((ext_vector_type(16))) _Float16 v16h;
typedef __attribute__((ext_vector_type(8)))  float    v8f;
typedef __attribute__((ext_vector_type(8)))  _Float16 h8;
typedef __attribute__((ext_vector_type(4)))  unsigned int u32x4;
typedef __attribute__((ext_vector_type(8)))  int i32x8;
typedef __attribute__((ext_vector_type(4)))  int i32x4;

#define BB 4
#define CC 64
#define DWC 128
#define HH 256
#define WWD 256
#define HWN 65536
#define RR 4
#define NE 3
#define SCALING 2.0f
#define EPSV 1e-6f

// workspace layout (bytes)
#define OFF_W1F   0u           // B*8192 f16  (conv1 weights, B-frag layout)
#define OFF_W3F   65536u       // B*4096 f16
#define OFF_W4F   98304u       // B*8192 f16
#define OFF_W5F   163840u      // B*4096 f16
#define OFF_W2E   196608u      // B*128*9 f32 (depthwise)
#define OFF_WSCA  215040u      // B*64*64 f32
#define OFF_PART  280576u      // B*64*256 f32 pooling partials
#define OFF_W3FS  542720u      // B*4096 f16 (conv3 weights with SCA scale folded)
#define OFF_T1    575488u      // B*128*65536 f16
#define OFF_T2    67684352u    // B*64*65536 f16 (end ~101.2 MB)

__device__ __forceinline__ int kmapA(int i, int lane) {
  // 16-bit A 16x32 fragment: element i of v16h -> K within 32-chunk
  return i + ((lane & 16) ? 16 : 0);
}

__device__ __forceinline__ v8f wmma32(v16h a, v16h b, v8f c) {
  return __builtin_amdgcn_wmma_f32_16x16x32_f16(false, a, false, b, (short)0, c, false, false);
}

// Tensor Data Mover: 2D tile load global->LDS (D# groups per CDNA5 ISA)
__device__ __forceinline__ void tdm_load2d(u32x4 g0, i32x8 g1) {
  i32x4 z4 = { 0, 0, 0, 0 };
#if defined(__clang_major__) && (__clang_major__ >= 23)
  i32x8 z8 = { 0, 0, 0, 0, 0, 0, 0, 0 };
  __builtin_amdgcn_tensor_load_to_lds(g0, g1, z4, z4, z8, 0);
#else
  __builtin_amdgcn_tensor_load_to_lds(g0, g1, z4, z4, 0);
#endif
}

// Build D# for: tile 16 px (contiguous, dim0) x 64 ch (stride HWN elems, dim1)
// dsz_log2: 1 = f16, 2 = f32.  pad: +1 DWORD per row in LDS.
__device__ __forceinline__ void tdm_tile16x64(unsigned lds_off, unsigned long long ga,
                                              unsigned dsz_log2, unsigned pad_interval) {
  u32x4 g0;
  g0.x = 1u;                                   // count=1, user mode
  g0.y = lds_off;                              // lds_addr
  g0.z = (unsigned)ga;                         // global_addr[31:0]
  g0.w = (unsigned)(ga >> 32) | 0x80000000u;   // global_addr[56:32] | type=2<<30
  i32x8 g1;
  g1[0] = (int)((dsz_log2 << 16) | (1u << 20) | (pad_interval << 22)); // data_size|pad_en|interval (amount=1 DW)
  g1[1] = (int)0xFFFF0000u;                    // tensor_dim0 lo16 (0x7FFFFFFF) @ [63:48]
  g1[2] = (int)0xFFFF7FFFu;                    // tensor_dim0 hi16 | tensor_dim1 lo16
  g1[3] = (int)((16u << 16) | 0x7FFFu);        // tensor_dim1 hi16 | tile_dim0=16
  g1[4] = 64;                                  // tile_dim1=64, tile_dim2=0
  g1[5] = HWN;                                 // tensor_dim0_stride lo32 (elements)
  g1[6] = 0;                                   // stride hi16 | dim1_stride lo16
  g1[7] = 0;
  tdm_load2d(g0, g1);
}

// ---------------------------------------------------------------------------
// K1: gate (top-1) + effective weights (LoRA folded), pre-swizzled for WMMA
// ---------------------------------------------------------------------------
__global__ void __launch_bounds__(256) k_prep(
    const float* __restrict__ probs,
    const float* __restrict__ w1, const float* __restrict__ la1, const float* __restrict__ lb1,
    const float* __restrict__ w2, const float* __restrict__ la2, const float* __restrict__ lb2,
    const float* __restrict__ wsca, const float* __restrict__ lasca, const float* __restrict__ lbsca,
    const float* __restrict__ w3, const float* __restrict__ la3, const float* __restrict__ lb3,
    const float* __restrict__ w4, const float* __restrict__ la4, const float* __restrict__ lb4,
    const float* __restrict__ w5, const float* __restrict__ la5, const float* __restrict__ lb5,
    unsigned char* __restrict__ ws)
{
  const int b = blockIdx.x;
  const int tid = threadIdx.x;

  float p0 = probs[b * NE + 0], p1 = probs[b * NE + 1], p2 = probs[b * NE + 2];
  int e = 0; float g = p0;
  if (p1 > g) { g = p1; e = 1; }
  if (p2 > g) { g = p2; e = 2; }
  const float gs = g * SCALING;

  _Float16* w1f = (_Float16*)(ws + OFF_W1F) + b * 8192;
  _Float16* w3f = (_Float16*)(ws + OFF_W3F) + b * 4096;
  _Float16* w4f = (_Float16*)(ws + OFF_W4F) + b * 8192;
  _Float16* w5f = (_Float16*)(ws + OFF_W5F) + b * 4096;
  float*    w2e = (float*)(ws + OFF_W2E) + b * DWC * 9;
  float*    wse = (float*)(ws + OFF_WSCA) + b * 4096;

  for (int idx = tid; idx < 8192; idx += 256) {
    int i = idx & 15, lane = (idx >> 4) & 31, kc = (idx >> 9) & 1, nt = idx >> 10;
    int oc = nt * 16 + (lane & 15);
    int k  = kc * 32 + i + ((lane & 16) ? 16 : 0);
    float acc = w1[oc * CC + k];
    for (int r = 0; r < RR; ++r)
      acc += gs * lb1[(e * DWC + oc) * RR + r] * la1[(e * RR + r) * CC + k];
    w1f[idx] = (_Float16)acc;
  }
  for (int idx = tid; idx < 4096; idx += 256) {
    int i = idx & 15, lane = (idx >> 4) & 31, kc = (idx >> 9) & 1, nt = idx >> 10;
    int oc = nt * 16 + (lane & 15);
    int k  = kc * 32 + i + ((lane & 16) ? 16 : 0);
    float acc = w3[oc * CC + k];
    for (int r = 0; r < RR; ++r)
      acc += gs * lb3[(e * CC + oc) * RR + r] * la3[(e * RR + r) * CC + k];
    w3f[idx] = (_Float16)acc;
  }
  for (int idx = tid; idx < 8192; idx += 256) {
    int i = idx & 15, lane = (idx >> 4) & 31, kc = (idx >> 9) & 1, nt = idx >> 10;
    int oc = nt * 16 + (lane & 15);
    int k  = kc * 32 + i + ((lane & 16) ? 16 : 0);
    float acc = w4[oc * CC + k];
    for (int r = 0; r < RR; ++r)
      acc += gs * lb4[(e * DWC + oc) * RR + r] * la4[(e * RR + r) * CC + k];
    w4f[idx] = (_Float16)acc;
  }
  for (int idx = tid; idx < 4096; idx += 256) {
    int i = idx & 15, lane = (idx >> 4) & 31, kc = (idx >> 9) & 1, nt = idx >> 10;
    int oc = nt * 16 + (lane & 15);
    int k  = kc * 32 + i + ((lane & 16) ? 16 : 0);
    float acc = w5[oc * CC + k];
    for (int r = 0; r < RR; ++r)
      acc += gs * lb5[(e * CC + oc) * RR + r] * la5[(e * RR + r) * CC + k];
    w5f[idx] = (_Float16)acc;
  }
  for (int idx = tid; idx < DWC * 9; idx += 256) {
    int oc = idx / 9, j = idx % 9, kh = j / 3, kw = j % 3;
    float acc = w2[oc * 9 + j];
    for (int r = 0; r < 12; ++r)
      acc += gs * lb2[(e * 384 + oc * 3 + kh) * 12 + r] * la2[(e * 12 + r) * 3 + kw];
    w2e[idx] = acc;
  }
  for (int idx = tid; idx < 4096; idx += 256) {
    int oc = idx >> 6, k = idx & 63;
    float acc = wsca[oc * 64 + k];
    for (int r = 0; r < RR; ++r)
      acc += gs * lbsca[(e * 64 + oc) * RR + r] * lasca[(e * RR + r) * 64 + k];
    wse[idx] = acc;
  }
}

// ---------------------------------------------------------------------------
// K2: fused LN1 + conv1 (1x1, 64->128) via WMMA; TDM stages input tile.
// grid: 4096 blocks x 128 threads; each wave = 16-pixel tile; LDS is [ch][px]
// ---------------------------------------------------------------------------
__global__ void __launch_bounds__(128) k_head(
    const float* __restrict__ inp, const float* __restrict__ ln1w,
    const float* __restrict__ ln1b, const float* __restrict__ b1,
    unsigned char* __restrict__ ws)
{
  __shared__ float sxT[4][64][17];   // TDM tile: [wave][channel][pixel + 1 pad DW]
  __shared__ float sd[4][16][17];
  __shared__ float s_lnw[CC], s_lnb[CC], s_b1[DWC];

  const int tid = threadIdx.x;
  const int wv = tid >> 5, lane = tid & 31;
  const int tile = blockIdx.x * 4 + wv;
  const int b = tile >> 12;              // 4096 tiles per image
  const int pixBase = (tile & 4095) << 4;
  const int m = lane & 15;

  // async tensor load: 16 px x 64 ch f32, rows padded to 17 DWORDs
  {
    unsigned lds_off = (unsigned)(unsigned long long)(const void*)&sxT[wv][0][0];
    unsigned long long ga =
        (unsigned long long)(const void*)(inp + (size_t)(b * CC) * HWN + pixBase);
    tdm_tile16x64(lds_off, ga, 2u /*4B*/, 3u /*16 DW rows*/);
  }
  if (tid < CC) { s_lnw[tid] = ln1w[tid]; s_lnb[tid] = ln1b[tid]; }
  if (tid < DWC) s_b1[tid] = b1[tid];

  const _Float16* w1f = (const _Float16*)(ws + OFF_W1F) + b * 8192;
  _Float16* t1 = (_Float16*)(ws + OFF_T1);

  __builtin_amdgcn_s_wait_tensorcnt(0);
  __syncthreads();

  // LayerNorm stats: lane pair (p, p+16) splits the 64 channels
  float s = 0.f, s2 = 0.f;
  {
    const int kk0 = (lane & 16) ? 32 : 0;
    for (int k = 0; k < 32; ++k) { float v = sxT[wv][kk0 + k][m]; s += v; s2 += v * v; }
    s  += __shfl_xor(s, 16);
    s2 += __shfl_xor(s2, 16);
  }
  const float mu = s * (1.0f / CC);
  const float rstd = rsqrtf(s2 * (1.0f / CC) - mu * mu + EPSV);

  v16h a0, a1;
#pragma unroll
  for (int i = 0; i < 16; ++i) {
    int k0 = kmapA(i, lane), k1 = 32 + k0;
    a0[i] = (_Float16)((sxT[wv][k0][m] - mu) * rstd * s_lnw[k0] + s_lnb[k0]);
    a1[i] = (_Float16)((sxT[wv][k1][m] - mu) * rstd * s_lnw[k1] + s_lnb[k1]);
  }

#pragma unroll
  for (int nt = 0; nt < 8; ++nt) {
    v16h wb0 = *(const v16h*)(w1f + ((nt * 2 + 0) * 32 + lane) * 16);
    v16h wb1 = *(const v16h*)(w1f + ((nt * 2 + 1) * 32 + lane) * 16);
    float bias = s_b1[nt * 16 + m];
    v8f acc = { bias, bias, bias, bias, bias, bias, bias, bias };
    acc = wmma32(a0, wb0, acc);
    acc = wmma32(a1, wb1, acc);

    __syncthreads();
#pragma unroll
    for (int r = 0; r < 8; ++r) {
      int mm = r + ((lane & 16) ? 8 : 0);
      sd[wv][mm][m] = acc[r];
    }
    __syncthreads();
    {
      int ocl = lane >> 1, part = lane & 1;
      h8 hv;
#pragma unroll
      for (int j = 0; j < 8; ++j) hv[j] = (_Float16)sd[wv][part * 8 + j][ocl];
      _Float16* dst = t1 + (size_t)(b * DWC + nt * 16 + ocl) * HWN + pixBase + part * 8;
      *(h8*)dst = hv;
    }
  }
}

// ---------------------------------------------------------------------------
// K3: 3x3 depthwise + SimpleGate -> t2 (f16); deterministic pooling partials
// ---------------------------------------------------------------------------
__global__ void __launch_bounds__(256) k_dw(const float* __restrict__ b2,
                                            unsigned char* __restrict__ ws)
{
  __shared__ float red[256];
  const int tid = threadIdx.x;
  const int blk = blockIdx.x;
  const int t = blk & 255;
  const int c = (blk >> 8) & 63;
  const int b = blk >> 14;
  const int pix = t * 256 + tid;
  const int h = pix >> 8, w = pix & 255;

  const float* w2e = (const float*)(ws + OFF_W2E) + b * DWC * 9;
  const _Float16* t1 = (const _Float16*)(ws + OFF_T1);
  _Float16* t2 = (_Float16*)(ws + OFF_T2);
  const _Float16* x1p = t1 + (size_t)(b * DWC + c) * HWN;
  const _Float16* x2p = t1 + (size_t)(b * DWC + c + 64) * HWN;

  float v1 = b2[c], v2 = b2[c + 64];
#pragma unroll
  for (int kh = 0; kh < 3; ++kh) {
    int h2 = h + kh - 1;
    if (h2 < 0 || h2 >= HH) continue;
#pragma unroll
    for (int kw = 0; kw < 3; ++kw) {
      int w2c = w + kw - 1;
      if (w2c < 0 || w2c >= WWD) continue;
      int np = h2 * WWD + w2c;
      v1 += w2e[c * 9 + kh * 3 + kw] * (float)x1p[np];
      v2 += w2e[(c + 64) * 9 + kh * 3 + kw] * (float)x2p[np];
    }
  }
  float xv = v1 * v2;                               // SimpleGate
  t2[(size_t)(b * CC + c) * HWN + pix] = (_Float16)xv;

  red[tid] = xv;
  __syncthreads();
  for (int sN = 128; sN > 0; sN >>= 1) {
    if (tid < sN) red[tid] += red[tid + sN];
    __syncthreads();
  }
  if (tid == 0)
    ((float*)(ws + OFF_PART))[(size_t)(b * CC + c) * 256 + t] = red[0];
}

// ---------------------------------------------------------------------------
// K4: deterministic pooled mean + SCA 1x1 -> scale, folded into conv3 weights
// ---------------------------------------------------------------------------
__global__ void __launch_bounds__(256) k_sca(const float* __restrict__ bsca,
                                             unsigned char* __restrict__ ws)
{
  __shared__ float sp[BB * CC];
  __shared__ float ssc[BB * CC];
  const int tid = threadIdx.x;
  const int b = tid >> 6, c = tid & 63;
  const float* part = (const float*)(ws + OFF_PART) + (size_t)(b * CC + c) * 256;
  float s = 0.f;
  for (int t = 0; t < 256; ++t) s += part[t];
  sp[tid] = s * (1.0f / (float)HWN);
  __syncthreads();
  const float* wse = (const float*)(ws + OFF_WSCA) + b * 4096 + c * 64;
  float v = bsca[c];
  for (int j = 0; j < 64; ++j) v += wse[j] * sp[b * CC + j];
  ssc[tid] = v;
  __syncthreads();
  // fold per-channel SCA scale into conv3 weight columns: w3fs = w3f * scale[b][k]
  const _Float16* src = (const _Float16*)(ws + OFF_W3F);
  _Float16* dst = (_Float16*)(ws + OFF_W3FS);
  for (int idx = tid; idx < BB * 4096; idx += 256) {
    int bb = idx >> 12, j = idx & 4095;
    int i = j & 15, lane = (j >> 4) & 31, kc = (j >> 9) & 1;
    int k = kc * 32 + i + ((lane & 16) ? 16 : 0);
    dst[idx] = (_Float16)((float)src[idx] * ssc[bb * CC + k]);
  }
}

// ---------------------------------------------------------------------------
// K5: TDM-staged fused tail: conv3 -> +inp*beta -> LN2 -> conv4 -> gate ->
//     conv5 -> out.  LDS is [ch][px]; 3 WMMA GEMMs per 16-pixel wave tile.
// ---------------------------------------------------------------------------
__global__ void __launch_bounds__(128) k_tail(
    const float* __restrict__ inp, const float* __restrict__ ln2w,
    const float* __restrict__ ln2b, const float* __restrict__ b3,
    const float* __restrict__ b4, const float* __restrict__ b5,
    const float* __restrict__ beta, const float* __restrict__ gamma,
    float* __restrict__ out, unsigned char* __restrict__ ws)
{
  __shared__ float    syT[4][64][17];    // inp tile (TDM), becomes y
  __shared__ _Float16 st2[4][64][18];    // t2 tile (TDM), rows padded to 18 halfs
  __shared__ float    sgT[4][128][17];   // conv4 out; rows 0..63 reused for out staging
  __shared__ float s_ln2w[CC], s_ln2b[CC], s_beta[CC], s_gamma[CC];
  __shared__ float s_b3[CC], s_b4[DWC], s_b5[CC];

  const int tid = threadIdx.x;
  const int wv = tid >> 5, lane = tid & 31;
  const int tile = blockIdx.x * 4 + wv;
  const int b = tile >> 12;
  const int pixBase = (tile & 4095) << 4;
  const int m = lane & 15;

  // async tensor loads: inp f32 tile and t2 f16 tile
  {
    unsigned lo_y = (unsigned)(unsigned long long)(const void*)&syT[wv][0][0];
    unsigned long long ga_y =
        (unsigned long long)(const void*)(inp + (size_t)(b * CC) * HWN + pixBase);
    tdm_tile16x64(lo_y, ga_y, 2u /*4B*/, 3u /*16 DW rows*/);

    const _Float16* t2 = (const _Float16*)(ws + OFF_T2);
    unsigned lo_x = (unsigned)(unsigned long long)(const void*)&st2[wv][0][0];
    unsigned long long ga_x =
        (unsigned long long)(const void*)(t2 + (size_t)(b * CC) * HWN + pixBase);
    tdm_tile16x64(lo_x, ga_x, 1u /*2B*/, 2u /*8 DW rows*/);
  }
  if (tid < CC) {
    s_ln2w[tid] = ln2w[tid]; s_ln2b[tid] = ln2b[tid];
    s_beta[tid] = beta[tid]; s_gamma[tid] = gamma[tid];
    s_b3[tid] = b3[tid]; s_b5[tid] = b5[tid];
  }
  if (tid < DWC) s_b4[tid] = b4[tid];

  __builtin_amdgcn_s_wait_tensorcnt(0);
  __syncthreads();

  // ---- conv3 (64->64): A = t2 tile (scale pre-folded into weights) ----
  v16h a0, a1;
#pragma unroll
  for (int i = 0; i < 16; ++i) {
    int k0 = kmapA(i, lane);
    a0[i] = st2[wv][k0][m];
    a1[i] = st2[wv][32 + k0][m];
  }
  const _Float16* w3fs = (const _Float16*)(ws + OFF_W3FS) + b * 4096;
#pragma unroll
  for (int nt = 0; nt < 4; ++nt) {
    v16h wb0 = *(const v16h*)(w3fs + ((nt * 2 + 0) * 32 + lane) * 16);
    v16h wb1 = *(const v16h*)(w3fs + ((nt * 2 + 1) * 32 + lane) * 16);
    float bias = s_b3[nt * 16 + m];
    v8f acc = { bias, bias, bias, bias, bias, bias, bias, bias };
    acc = wmma32(a0, wb0, acc);
    acc = wmma32(a1, wb1, acc);
#pragma unroll
    for (int r = 0; r < 8; ++r) {
      int mm = r + ((lane & 16) ? 8 : 0);
      int ch = nt * 16 + m;
      syT[wv][ch][mm] += acc[r] * s_beta[ch];   // y = inp + x3*beta
    }
  }
  __syncthreads();

  // ---- LN2 on y ----
  float s = 0.f, s2 = 0.f;
  {
    const int kk0 = (lane & 16) ? 32 : 0;
    for (int k = 0; k < 32; ++k) { float v = syT[wv][kk0 + k][m]; s += v; s2 += v * v; }
    s  += __shfl_xor(s, 16);
    s2 += __shfl_xor(s2, 16);
  }
  const float mu2 = s * (1.0f / CC);
  const float rstd2 = rsqrtf(s2 * (1.0f / CC) - mu2 * mu2 + EPSV);

  v16h n0, n1;
#pragma unroll
  for (int i = 0; i < 16; ++i) {
    int k0 = kmapA(i, lane), k1 = 32 + k0;
    n0[i] = (_Float16)((syT[wv][k0][m] - mu2) * rstd2 * s_ln2w[k0] + s_ln2b[k0]);
    n1[i] = (_Float16)((syT[wv][k1][m] - mu2) * rstd2 * s_ln2w[k1] + s_ln2b[k1]);
  }

  // ---- conv4 (64->128) -> sgT[ch][px] ----
  const _Float16* w4f = (const _Float16*)(ws + OFF_W4F) + b * 8192;
#pragma unroll
  for (int nt = 0; nt < 8; ++nt) {
    v16h wb0 = *(const v16h*)(w4f + ((nt * 2 + 0) * 32 + lane) * 16);
    v16h wb1 = *(const v16h*)(w4f + ((nt * 2 + 1) * 32 + lane) * 16);
    float bias = s_b4[nt * 16 + m];
    v8f acc = { bias, bias, bias, bias, bias, bias, bias, bias };
    acc = wmma32(n0, wb0, acc);
    acc = wmma32(n1, wb1, acc);
#pragma unroll
    for (int r = 0; r < 8; ++r) {
      int mm = r + ((lane & 16) ? 8 : 0);
      sgT[wv][nt * 16 + m][mm] = acc[r];
    }
  }
  __syncthreads();

  // ---- SimpleGate + conv5 (64->64) ----
  v16h g0, g1;
#pragma unroll
  for (int i = 0; i < 16; ++i) {
    int k0 = kmapA(i, lane), k1 = 32 + k0;
    g0[i] = (_Float16)(sgT[wv][k0][m] * sgT[wv][64 + k0][m]);
    g1[i] = (_Float16)(sgT[wv][k1][m] * sgT[wv][64 + k1][m]);
  }
  const _Float16* w5f = (const _Float16*)(ws + OFF_W5F) + b * 4096;
#pragma unroll
  for (int nt = 0; nt < 4; ++nt) {
    v16h wb0 = *(const v16h*)(w5f + ((nt * 2 + 0) * 32 + lane) * 16);
    v16h wb1 = *(const v16h*)(w5f + ((nt * 2 + 1) * 32 + lane) * 16);
    float bias = s_b5[nt * 16 + m];
    v8f acc = { bias, bias, bias, bias, bias, bias, bias, bias };
    acc = wmma32(g0, wb0, acc);
    acc = wmma32(g1, wb1, acc);

    __syncthreads();
#pragma unroll
    for (int r = 0; r < 8; ++r) {
      int mm = r + ((lane & 16) ? 8 : 0);
      int ch = nt * 16 + m;
      sgT[wv][ch][mm] = syT[wv][ch][mm] + acc[r] * s_gamma[ch];  // out = y + x5*gamma
    }
    __syncthreads();
    {
      int ocl = lane >> 1, part = lane & 1;
      int ch = nt * 16 + ocl;
      float* dst = out + (size_t)(b * CC + ch) * HWN + pixBase + part * 8;
      float4 fa, fb;
      fa.x = sgT[wv][ch][part * 8 + 0]; fa.y = sgT[wv][ch][part * 8 + 1];
      fa.z = sgT[wv][ch][part * 8 + 2]; fa.w = sgT[wv][ch][part * 8 + 3];
      fb.x = sgT[wv][ch][part * 8 + 4]; fb.y = sgT[wv][ch][part * 8 + 5];
      fb.z = sgT[wv][ch][part * 8 + 6]; fb.w = sgT[wv][ch][part * 8 + 7];
      *(float4*)dst = fa;
      *(float4*)(dst + 4) = fb;
    }
  }
}

// ---------------------------------------------------------------------------
extern "C" void kernel_launch(void* const* d_in, const int* in_sizes, int n_in,
                              void* d_out, int out_size, void* d_ws, size_t ws_size,
                              hipStream_t stream)
{
  (void)in_sizes; (void)n_in; (void)out_size; (void)ws_size;
  const float* inp   = (const float*)d_in[0];
  const float* probs = (const float*)d_in[1];
  const float* ln1w  = (const float*)d_in[2];
  const float* ln1b  = (const float*)d_in[3];
  const float* ln2w  = (const float*)d_in[4];
  const float* ln2b  = (const float*)d_in[5];
  const float* w1    = (const float*)d_in[6];
  const float* b1    = (const float*)d_in[7];
  const float* la1   = (const float*)d_in[8];
  const float* lb1   = (const float*)d_in[9];
  const float* w2    = (const float*)d_in[10];
  const float* b2    = (const float*)d_in[11];
  const float* la2   = (const float*)d_in[12];
  const float* lb2   = (const float*)d_in[13];
  const float* wsca  = (const float*)d_in[14];
  const float* bsca  = (const float*)d_in[15];
  const float* lasca = (const float*)d_in[16];
  const float* lbsca = (const float*)d_in[17];
  const float* w3    = (const float*)d_in[18];
  const float* b3    = (const float*)d_in[19];
  const float* la3   = (const float*)d_in[20];
  const float* lb3   = (const float*)d_in[21];
  const float* w4    = (const float*)d_in[22];
  const float* b4    = (const float*)d_in[23];
  const float* la4   = (const float*)d_in[24];
  const float* lb4   = (const float*)d_in[25];
  const float* w5    = (const float*)d_in[26];
  const float* b5    = (const float*)d_in[27];
  const float* la5   = (const float*)d_in[28];
  const float* lb5   = (const float*)d_in[29];
  const float* beta  = (const float*)d_in[30];
  const float* gamma = (const float*)d_in[31];
  unsigned char* ws = (unsigned char*)d_ws;
  float* out = (float*)d_out;

  k_prep<<<BB, 256, 0, stream>>>(probs, w1, la1, lb1, w2, la2, lb2,
                                 wsca, lasca, lbsca, w3, la3, lb3,
                                 w4, la4, lb4, w5, la5, lb5, ws);
  k_head<<<4096, 128, 0, stream>>>(inp, ln1w, ln1b, b1, ws);
  k_dw<<<65536, 256, 0, stream>>>(b2, ws);
  k_sca<<<1, 256, 0, stream>>>(bsca, ws);
  k_tail<<<4096, 128, 0, stream>>>(inp, ln2w, ln2b, b3, b4, b5, beta, gamma, out, ws);
}
